// AxialShift_58265526337854
// MI455X (gfx1250) — matrix operand order repaired and checked by reading, not compiled, not run on hardware
//
#include <hip/hip_runtime.h>
#include <hip/hip_bf16.h>
#include <math.h>

#define C_   96
#define Hd   256
#define Wd   256
#define HW   65536              // 256*256
#define Bn   4
#define NPS  (C_ * HW)          // elements per sample = 6291456
#define ACTS 104                // padded LDS pixel stride (f16 elements), 16B aligned

typedef __attribute__((ext_vector_type(16))) _Float16 v16h;
typedef __attribute__((ext_vector_type(8)))  _Float16 v8h;
typedef __attribute__((ext_vector_type(4)))  _Float16 v4h;
typedef __attribute__((ext_vector_type(8)))  float    v8f;

__device__ __forceinline__ v16h cat8(v8h lo, v8h hi) {
    return __builtin_shufflevector(lo, hi, 0,1,2,3,4,5,6,7,8,9,10,11,12,13,14,15);
}
// B fragment: 16 contiguous halves (lane holds K=khalf*16 .. +15 for its column)
__device__ __forceinline__ v16h ldB(const _Float16* p) {
    return cat8(*(const v8h*)p, *(const v8h*)(p + 8));
}
// A fragment: per ISA layout, lane holds K blocks {kb..kb+7} and {kb+16..kb+23}
__device__ __forceinline__ v16h ldA(const _Float16* p) {
    return cat8(*(const v8h*)p, *(const v8h*)(p + 16));
}
__device__ __forceinline__ float gelu_erf(float v) {
    return 0.5f * v * (1.0f + erff(v * 0.70710678118654752f));
}
__device__ __forceinline__ void wave_reduce_atomic(float s, float sq, float* sum_p, float* sq_p) {
    #pragma unroll
    for (int off = 16; off; off >>= 1) {
        s  += __shfl_xor(s,  off);
        sq += __shfl_xor(sq, off);
    }
    if ((threadIdx.x & 31) == 0) { atomicAdd(sum_p, s); atomicAdd(sq_p, sq); }
}

// ---------------- stats helpers ----------------
__global__ void k_zero_stats(float* stats) {
    if (threadIdx.x < 32) stats[threadIdx.x] = 0.0f;
}
__global__ void k_finalize(float* stats, int base) {
    int b = threadIdx.x;
    if (b < Bn) {
        const float invN = 1.0f / (float)NPS;
        float m = stats[base + b] * invN;
        float v = stats[base + 4 + b] * invN - m * m;
        stats[base + 8  + b] = m;
        stats[base + 12 + b] = rsqrtf(v + 1e-5f);
    }
}

// ---------------- kernel 1: t = conv1(x)+b1 (f16 out) + GN1 stats ----------------
__global__ __launch_bounds__(128) void k_conv1(const float* __restrict__ x,
                                               const float* __restrict__ w1,
                                               const float* __restrict__ b1,
                                               _Float16* __restrict__ t16,
                                               float* __restrict__ stats) {
    __shared__ _Float16 wLds[C_ * C_];
    __shared__ float    bLds[C_];
    __shared__ _Float16 act[4][16 * ACTS];

    const int tid = threadIdx.x;
    for (int i = tid; i < C_ * C_; i += 128) wLds[i] = (_Float16)w1[i];
    for (int i = tid; i < C_; i += 128)      bLds[i] = b1[i];
    __syncthreads();

    const int blk = blockIdx.x;
    const int wt = blk & 3, h = (blk >> 2) & 255, b = blk >> 10;
    const int wave = tid >> 5, lane = tid & 31;
    const int p0 = wt * 64 + wave * 16;
    const int khalf = lane >> 4, nn = lane & 15;

    const float* xb = x + (size_t)b * NPS + (size_t)h * Wd + p0;
    _Float16* a = &act[wave][0];

    #pragma unroll
    for (int it = 0; it < 12; ++it) {
        const int c  = it * 8 + (lane >> 2);
        const int px = (lane & 3) * 4;
        if (it < 11) __builtin_prefetch(xb + (size_t)(c + 8) * HW + px, 0, 0);
        const float4 v = *(const float4*)(xb + (size_t)c * HW + px);
        _Float16* dst = a + px * ACTS + c;
        dst[0]        = (_Float16)v.x;
        dst[ACTS]     = (_Float16)v.y;
        dst[2 * ACTS] = (_Float16)v.z;
        dst[3 * ACTS] = (_Float16)v.w;
    }
    asm volatile("s_wait_dscnt 0" ::: "memory");

    v8f acc[6];
    #pragma unroll
    for (int m = 0; m < 6; ++m)
        #pragma unroll
        for (int j = 0; j < 8; ++j) acc[m][j] = 0.0f;

    #pragma unroll
    for (int ks = 0; ks < 3; ++ks) {
        const v16h bf = ldA(a + nn * ACTS + ks * 32 + khalf * 8);
        #pragma unroll
        for (int m = 0; m < 6; ++m) {
            const v16h af = ldA(wLds + (m * 16 + nn) * C_ + ks * 32 + khalf * 8);
            acc[m] = __builtin_amdgcn_wmma_f32_16x16x32_f16(false, af, false, bf,
                                                            (short)0, acc[m], false, false);
        }
    }

    float s = 0.0f, sq = 0.0f;
    const int pix = h * Wd + p0 + nn;
    _Float16* tb = t16 + (size_t)b * NPS + pix;
    #pragma unroll
    for (int m = 0; m < 6; ++m)
        #pragma unroll
        for (int j = 0; j < 8; ++j) {
            const int o = m * 16 + khalf * 8 + j;
            const float v = acc[m][j] + bLds[o];
            s += v; sq += v * v;
            tb[(size_t)o * HW] = (_Float16)v;
        }
    wave_reduce_atomic(s, sq, stats + b, stats + 4 + b);
}

// ---------------- kernel 2: y = f16(gelu(GN1(t))) ----------------
__global__ __launch_bounds__(256) void k_gn_gelu(const _Float16* __restrict__ t16,
                                                 _Float16* __restrict__ y16,
                                                 const float* __restrict__ stats,
                                                 const float* __restrict__ g,
                                                 const float* __restrict__ be) {
    const size_t i = (size_t)blockIdx.x * 256 + threadIdx.x;
    const size_t N = (size_t)Bn * NPS;
    if (i >= N) return;
    const int b = (int)(i / NPS);
    const int c = (int)((i / HW) % C_);
    const float m = stats[8 + b], r = stats[12 + b];
    const float v = ((float)t16[i] - m) * r * g[c] + be[c];
    y16[i] = (_Float16)gelu_erf(v);
}

// ---------------- kernel 3: 4 shifted branches + GN2 stats ----------------
__global__ __launch_bounds__(128) void k_branches(const _Float16* __restrict__ y16,
                                                  const float* __restrict__ w21,
                                                  const float* __restrict__ b21,
                                                  const float* __restrict__ w22,
                                                  const float* __restrict__ b22,
                                                  _Float16* __restrict__ z16,
                                                  float* __restrict__ stats) {
    __shared__ _Float16 wA[C_ * C_];
    __shared__ _Float16 wB[C_ * C_];
    __shared__ float    bA[C_], bB[C_];
    __shared__ _Float16 act[4][16 * ACTS];

    const int tid = threadIdx.x;
    for (int i = tid; i < C_ * C_; i += 128) { wA[i] = (_Float16)w21[i]; wB[i] = (_Float16)w22[i]; }
    for (int i = tid; i < C_; i += 128)      { bA[i] = b21[i];           bB[i] = b22[i]; }
    __syncthreads();

    const int blk = blockIdx.x;
    const int wt = blk & 3, h = (blk >> 2) & 255, b = blk >> 10;
    const int wave = tid >> 5, lane = tid & 31;
    const int p0 = wt * 64 + wave * 16;
    const int khalf = lane >> 4, nn = lane & 15;

    const _Float16* yb = y16 + (size_t)b * NPS;
    _Float16* a = &act[wave][0];

    float zsum[48];
    #pragma unroll
    for (int q = 0; q < 48; ++q) zsum[q] = 0.0f;

    // branch source pixel = (h - ah*s, w - aw*s), s = c/32 - 1
    const int AH[4] = {0, 1, 1, 1};   // lr, l_diag, td, r_diag
    const int AW[4] = {1, 1, 0, -1};

    #pragma unroll
    for (int br = 0; br < 4; ++br) {
        #pragma unroll
        for (int it = 0; it < 12; ++it) {
            const int c  = it * 8 + (lane >> 2);
            const int s  = c / 32 - 1;
            const int hs = h - AH[br] * s;
            const int wbase = p0 + (lane & 3) * 4 - AW[br] * s;
            _Float16* dst = a + (lane & 3) * 4 * ACTS + c;
            const _Float16* src = yb + (size_t)c * HW + (size_t)hs * Wd;
            #pragma unroll
            for (int e = 0; e < 4; ++e) {
                const int wc = wbase + e;
                _Float16 v = (_Float16)0.0f;
                if ((unsigned)hs < (unsigned)Hd && (unsigned)wc < (unsigned)Wd) v = src[wc];
                dst[e * ACTS] = v;
            }
        }
        asm volatile("s_wait_dscnt 0" ::: "memory");

        const _Float16* wS = (br < 2) ? wA : wB;
        const float*    bS = (br < 2) ? bA : bB;

        v8f acc[6];
        #pragma unroll
        for (int m = 0; m < 6; ++m)
            #pragma unroll
            for (int j = 0; j < 8; ++j) acc[m][j] = 0.0f;

        #pragma unroll
        for (int ks = 0; ks < 3; ++ks) {
            const v16h bf = ldA(a + nn * ACTS + ks * 32 + khalf * 8);
            #pragma unroll
            for (int m = 0; m < 6; ++m) {
                const v16h af = ldA(wS + (m * 16 + nn) * C_ + ks * 32 + khalf * 8);
                acc[m] = __builtin_amdgcn_wmma_f32_16x16x32_f16(false, af, false, bf,
                                                                (short)0, acc[m], false, false);
            }
        }
        #pragma unroll
        for (int m = 0; m < 6; ++m)
            #pragma unroll
            for (int j = 0; j < 8; ++j) {
                const int o = m * 16 + khalf * 8 + j;
                zsum[m * 8 + j] += gelu_erf(acc[m][j] + bS[o]);
            }
        // DS ops from one wave complete in order: next branch's stores are WAR-safe.
    }

    float s = 0.0f, sq = 0.0f;
    const int pix = h * Wd + p0 + nn;
    _Float16* zb = z16 + (size_t)b * NPS + pix;
    #pragma unroll
    for (int m = 0; m < 6; ++m)
        #pragma unroll
        for (int j = 0; j < 8; ++j) {
            const int o = m * 16 + khalf * 8 + j;
            const float v = zsum[m * 8 + j];
            s += v; sq += v * v;
            zb[(size_t)o * HW] = (_Float16)v;
        }
    wave_reduce_atomic(s, sq, stats + 16 + b, stats + 20 + b);
}

// ---------------- kernel 4: out = conv3(GN2(z)) + b3 ----------------
__global__ __launch_bounds__(128) void k_final(const _Float16* __restrict__ z16,
                                               const float* __restrict__ w3,
                                               const float* __restrict__ b3,
                                               const float* __restrict__ g2,
                                               const float* __restrict__ be2,
                                               const float* __restrict__ stats,
                                               float* __restrict__ out) {
    __shared__ _Float16 wLds[C_ * C_];
    __shared__ float    bLds[C_], gm[C_], bt[C_];
    __shared__ _Float16 act[4][16 * ACTS];

    const int tid = threadIdx.x;
    for (int i = tid; i < C_ * C_; i += 128) wLds[i] = (_Float16)w3[i];
    for (int i = tid; i < C_; i += 128) { bLds[i] = b3[i]; gm[i] = g2[i]; bt[i] = be2[i]; }
    __syncthreads();

    const int blk = blockIdx.x;
    const int wt = blk & 3, h = (blk >> 2) & 255, b = blk >> 10;
    const int wave = tid >> 5, lane = tid & 31;
    const int p0 = wt * 64 + wave * 16;
    const int khalf = lane >> 4, nn = lane & 15;

    const float mean = stats[24 + b], r = stats[28 + b];
    const _Float16* zb = z16 + (size_t)b * NPS + (size_t)h * Wd + p0;
    _Float16* a = &act[wave][0];

    #pragma unroll
    for (int it = 0; it < 12; ++it) {
        const int c  = it * 8 + (lane >> 2);
        const int px = (lane & 3) * 4;
        const float sc = r * gm[c], of = bt[c] - mean * sc;
        const v4h zv = *(const v4h*)(zb + (size_t)c * HW + px);
        _Float16* dst = a + px * ACTS + c;
        #pragma unroll
        for (int e = 0; e < 4; ++e)
            dst[e * ACTS] = (_Float16)((float)zv[e] * sc + of);
    }
    asm volatile("s_wait_dscnt 0" ::: "memory");

    v8f acc[6];
    #pragma unroll
    for (int m = 0; m < 6; ++m)
        #pragma unroll
        for (int j = 0; j < 8; ++j) acc[m][j] = 0.0f;

    #pragma unroll
    for (int ks = 0; ks < 3; ++ks) {
        const v16h bf = ldA(a + nn * ACTS + ks * 32 + khalf * 8);
        #pragma unroll
        for (int m = 0; m < 6; ++m) {
            const v16h af = ldA(wLds + (m * 16 + nn) * C_ + ks * 32 + khalf * 8);
            acc[m] = __builtin_amdgcn_wmma_f32_16x16x32_f16(false, af, false, bf,
                                                            (short)0, acc[m], false, false);
        }
    }

    const int pix = h * Wd + p0 + nn;
    float* ob = out + (size_t)b * NPS + pix;
    #pragma unroll
    for (int m = 0; m < 6; ++m)
        #pragma unroll
        for (int j = 0; j < 8; ++j) {
            const int o = m * 16 + khalf * 8 + j;
            ob[(size_t)o * HW] = acc[m][j] + bLds[o];
        }
}

// ---------------- launcher ----------------
extern "C" void kernel_launch(void* const* d_in, const int* in_sizes, int n_in,
                              void* d_out, int out_size, void* d_ws, size_t ws_size,
                              hipStream_t stream) {
    const float* x     = (const float*)d_in[0];
    const float* w1    = (const float*)d_in[1];
    const float* b1    = (const float*)d_in[2];
    const float* w21   = (const float*)d_in[3];
    const float* b21   = (const float*)d_in[4];
    const float* w22   = (const float*)d_in[5];
    const float* b22   = (const float*)d_in[6];
    const float* w3    = (const float*)d_in[7];
    const float* b3    = (const float*)d_in[8];
    const float* gn1_w = (const float*)d_in[9];
    const float* gn1_b = (const float*)d_in[10];
    const float* gn2_w = (const float*)d_in[11];
    const float* gn2_b = (const float*)d_in[12];
    float* out = (float*)d_out;

    float*    stats = (float*)d_ws;                                   // 32 floats used
    _Float16* t16   = (_Float16*)((char*)d_ws + 256);                 // [4,96,256,256] f16
    _Float16* y16   = (_Float16*)((char*)d_ws + 256 + (size_t)Bn * NPS * 2);
    _Float16* z16   = t16;                                            // reuse (t dead after k2)

    const int tiles = Bn * Hd * 4;                                    // 4096 blocks, 64 px each
    const size_t Nel = (size_t)Bn * NPS;                              // 25165824
    const int ew_blocks = (int)((Nel + 255) / 256);

    k_zero_stats<<<1, 32, 0, stream>>>(stats);
    k_conv1<<<tiles, 128, 0, stream>>>(x, w1, b1, t16, stats);
    k_finalize<<<1, 4, 0, stream>>>(stats, 0);
    k_gn_gelu<<<ew_blocks, 256, 0, stream>>>(t16, y16, stats, gn1_w, gn1_b);
    k_branches<<<tiles, 128, 0, stream>>>(y16, w21, b21, w22, b22, z16, stats);
    k_finalize<<<1, 4, 0, stream>>>(stats, 16);
    k_final<<<tiles, 128, 0, stream>>>(z16, w3, b3, gn2_w, gn2_b, stats, out);
}
